// GQAAttention_16793322127754
// MI455X (gfx1250) — matrix operand order repaired and checked by reading, not compiled, not run on hardware
//
#include <hip/hip_runtime.h>
#include <hip/hip_bf16.h>
#include <math.h>

// ---------------------------------------------------------------------------
// Problem constants (from reference)
// ---------------------------------------------------------------------------
#define DIMV      2048
#define NHEADS    32
#define NKVHEADS  8
#define HEADD     64          // DIM / NUM_HEADS
#define KVD       512         // HEAD_DIM * NUM_KV_HEADS
#define BSZ_      2
#define SEQ_      2048
#define MROWS     (BSZ_ * SEQ_)   // 4096 flattened rows
#define THRESH    0.05f
#define LOG2E     1.44269504088896f

typedef _Float16 half8 __attribute__((ext_vector_type(8)));
typedef __attribute__((ext_vector_type(16))) _Float16 v16h;
typedef __attribute__((ext_vector_type(8)))  float    v8f;

// ---------------------------------------------------------------------------
// Utility kernels: zero, deterministic two-stage mean(|w|), ternarize, cvt
// ---------------------------------------------------------------------------
__global__ void k_zero_f32(float* __restrict__ p, int n) {
  int i = blockIdx.x * blockDim.x + threadIdx.x;
  if (i < n) p[i] = 0.0f;
}

__global__ void k_abs_sum_partial(const float* __restrict__ w, int n,
                                  float* __restrict__ partial) {
  __shared__ float sm[256];
  float s = 0.0f;
  for (int i = blockIdx.x * blockDim.x + threadIdx.x; i < n;
       i += gridDim.x * blockDim.x)
    s += fabsf(w[i]);
  sm[threadIdx.x] = s;
  __syncthreads();
  for (int st = 128; st > 0; st >>= 1) {
    if (threadIdx.x < st) sm[threadIdx.x] += sm[threadIdx.x + st];
    __syncthreads();
  }
  if (threadIdx.x == 0) partial[blockIdx.x] = sm[0];
}

__global__ void k_finalize_scale(const float* __restrict__ partial, int nblocks,
                                 float inv_n, float* __restrict__ scale_out) {
  __shared__ float sm[256];
  float s = 0.0f;
  for (int i = threadIdx.x; i < nblocks; i += 256) s += partial[i];
  sm[threadIdx.x] = s;
  __syncthreads();
  for (int st = 128; st > 0; st >>= 1) {
    if (threadIdx.x < st) sm[threadIdx.x] += sm[threadIdx.x + st];
    __syncthreads();
  }
  if (threadIdx.x == 0) *scale_out = fmaxf(sm[0] * inv_n, 1e-6f);
}

// ternary {-1,0,1} is exact in fp16
__global__ void k_ternarize_f16(const float* __restrict__ w,
                                _Float16* __restrict__ out, int n,
                                const float* __restrict__ scale) {
  const float thr = THRESH * (*scale);
  for (int i = blockIdx.x * blockDim.x + threadIdx.x; i < n;
       i += gridDim.x * blockDim.x) {
    float v = w[i];
    float t = (v > thr) ? 1.0f : ((v < -thr) ? -1.0f : 0.0f);
    out[i] = (_Float16)t;
  }
}

__global__ void k_cvt_f16(const float* __restrict__ x,
                          _Float16* __restrict__ out, int n) {
  for (int i = blockIdx.x * blockDim.x + threadIdx.x; i < n;
       i += gridDim.x * blockDim.x)
    out[i] = (_Float16)x[i];
}

// ---------------------------------------------------------------------------
// WMMA GEMM:  C[M][N] = A[M][K] @ W[N][K]^T   (F.linear semantics)
// blockDim = 128 (4 waves). Each wave computes a 16(M) x 64(N) strip:
// A-fragment loaded once per K-step of 32, reused across 4 N-tiles.
// Fragment layouts: contiguous k = 16*hf + j in both A and B (wave32, 16-bit operands).
// ---------------------------------------------------------------------------
template <bool OUT_F32>
__global__ void k_gemm_wmma(const _Float16* __restrict__ A,
                            const _Float16* __restrict__ W,
                            void* __restrict__ Cout,
                            int M, int N, int K) {
  const int lane = threadIdx.x & 31;
  const int wave = threadIdx.x >> 5;       // 0..3
  const int hf   = lane >> 4;              // half-wave select
  const int l16  = lane & 15;

  const int nbase = blockIdx.x * 64;       // 4 N-tiles of 16
  const int mtile = blockIdx.y * 4 + wave; // one 16-row M tile per wave
  const int mrow  = mtile * 16 + l16;      // A row owned by this lane
  const int ncol  = nbase + l16;           // base B column owned by this lane

  v8f acc[4] = {};
  const _Float16* arow = A + (size_t)mrow * K;

  for (int kb = 0; kb < K; kb += 32) {
    if (kb + 256 < K) __builtin_prefetch(arow + kb + 256, 0, 1);
    // A fragment: a[j] = A[mrow][kb + hf*16 + j]
    half8 alo = *(const half8*)(arow + kb + hf * 16);
    half8 ahi = *(const half8*)(arow + kb + hf * 16 + 8);
    v16h a;
#pragma unroll
    for (int i = 0; i < 8; ++i) { a[i] = alo[i]; a[i + 8] = ahi[i]; }

#pragma unroll
    for (int t = 0; t < 4; ++t) {
      // B fragment: b[j] = W[ncol + 16t][kb + hf*16 + j] (contiguous run)
      const _Float16* bp = W + (size_t)(ncol + t * 16) * K + kb + hf * 16;
      half8 blo = *(const half8*)(bp);
      half8 bhi = *(const half8*)(bp + 8);
      v16h b;
#pragma unroll
      for (int i = 0; i < 8; ++i) { b[i] = blo[i]; b[i + 8] = bhi[i]; }
      acc[t] = __builtin_amdgcn_wmma_f32_16x16x32_f16(
          false, a, false, b, (short)0, acc[t], false, false);
    }
  }

  // C layout: reg r -> row = 16*mtile + r + 8*hf, col = nbase + 16t + l16
#pragma unroll
  for (int t = 0; t < 4; ++t) {
#pragma unroll
    for (int r = 0; r < 8; ++r) {
      size_t off = (size_t)(mtile * 16 + r + hf * 8) * N + nbase + t * 16 + l16;
      if (OUT_F32) ((float*)Cout)[off] = acc[t][r];
      else         ((_Float16*)Cout)[off] = (_Float16)acc[t][r];
    }
  }
}

// ---------------------------------------------------------------------------
// Flash-style causal GQA attention, one wave (32 threads) per 16-query tile.
//   q:  [B*S][2048]  (head h at col h*64)
//   k,v:[B*S][512]   (kv-head h/4 at col (h/4)*64)
// Streams 32-key blocks. S^T = K @ Q^T via WMMA (rows = keys, cols = queries)
// so the softmax reduction over keys is in-lane + one xor-16 shuffle.
// P is staged through LDS to re-layout it as the A operand of P @ V.
// ---------------------------------------------------------------------------
__global__ void k_attn_wmma(const _Float16* __restrict__ qh,
                            const _Float16* __restrict__ kh,
                            const _Float16* __restrict__ vh,
                            _Float16* __restrict__ oh) {
  __shared__ _Float16 Pt[16][32];          // P^T staging: [q][k-in-block]

  const int lane  = threadIdx.x;           // blockDim.x == 32
  const int hf    = lane >> 4;
  const int l16   = lane & 15;
  const int bh    = blockIdx.y;            // b*32 + h
  const int b     = bh >> 5;
  const int h     = bh & 31;
  const int kvh   = h >> 2;                // KV_REPEAT = 4
  const int qtile = blockIdx.x;            // 16 queries per tile
  const size_t rowbase = (size_t)b * SEQ_;
  const int qcoff = h * HEADD;             // offset inside width-2048 row
  const int kcoff = kvh * HEADD;           // offset inside width-512 row

  // Q as B-operand of S^T: b[j] = Q[qtile*16 + l16][ds*32 + hf*16 + j]
  v16h qb[2];
#pragma unroll
  for (int ds = 0; ds < 2; ++ds) {
    const _Float16* p =
        qh + (rowbase + qtile * 16 + l16) * DIMV + qcoff + ds * 32 + hf * 16;
    half8 lo = *(const half8*)p;
    half8 hi = *(const half8*)(p + 8);
#pragma unroll
    for (int i = 0; i < 8; ++i) { qb[ds][i] = lo[i]; qb[ds][i + 8] = hi[i]; }
  }

  v8f   o[4] = {};                 // O tiles [q][d], 4 d-tiles of 16
  float m_run = -__builtin_inff(); // running max  (column layout: q = l16)
  float l_run = 0.0f;              // running sum

  const int nkb = (qtile >> 1) + 1;        // 32-key blocks needed (causal)
  for (int kb = 0; kb < nkb; ++kb) {
    const int kbase = kb * 32;

    // ---- S^T tiles: st[kt] covers key rows kbase+16*kt .. +15 ----
    v8f st[2] = {};
#pragma unroll
    for (int kt = 0; kt < 2; ++kt) {
      const _Float16* krp =
          kh + (rowbase + kbase + kt * 16 + l16) * KVD + kcoff;
#pragma unroll
      for (int ds = 0; ds < 2; ++ds) {
        half8 alo = *(const half8*)(krp + ds * 32 + hf * 16);
        half8 ahi = *(const half8*)(krp + ds * 32 + hf * 16 + 8);
        v16h a;
#pragma unroll
        for (int i = 0; i < 8; ++i) { a[i] = alo[i]; a[i + 8] = ahi[i]; }
        st[kt] = __builtin_amdgcn_wmma_f32_16x16x32_f16(
            false, a, false, qb[ds], (short)0, st[kt], false, false);
      }
    }

    // ---- scale + causal mask + online softmax (per query column) ----
    const int qg = qtile * 16 + l16;       // this lane's query column
    float sv[16];
    float mx = -__builtin_inff();
#pragma unroll
    for (int kt = 0; kt < 2; ++kt) {
#pragma unroll
      for (int r = 0; r < 8; ++r) {
        int kg = kbase + kt * 16 + r + hf * 8;   // key row of this element
        float v = st[kt][r] * 0.125f;            // 1/sqrt(64)
        v = (kg <= qg) ? v : -__builtin_inff();
        sv[kt * 8 + r] = v;
        mx = fmaxf(mx, v);
      }
    }
    mx = fmaxf(mx, __shfl_xor(mx, 16, 32));      // full column max
    float m_new = fmaxf(m_run, mx);
    float alpha = exp2f((m_run - m_new) * LOG2E);
    float lsum = 0.0f;
    _Float16 pv[16];
#pragma unroll
    for (int i = 0; i < 16; ++i) {
      float p = exp2f((sv[i] - m_new) * LOG2E);
      lsum += p;
      pv[i] = (_Float16)p;
    }
    lsum += __shfl_xor(lsum, 16, 32);
    l_run = l_run * alpha + lsum;
    m_run = m_new;

    // ---- stage P^T -> LDS: Pt[q][k-in-block] ----
#pragma unroll
    for (int kt = 0; kt < 2; ++kt)
#pragma unroll
      for (int r = 0; r < 8; ++r)
        Pt[l16][kt * 16 + r + hf * 8] = pv[kt * 8 + r];
    __syncthreads();

    // ---- rescale O (per query ROW now): fetch alpha for rows r+8*hf ----
    float arw[8];
#pragma unroll
    for (int r = 0; r < 8; ++r) arw[r] = __shfl(alpha, hf * 8 + r, 32);
#pragma unroll
    for (int t = 0; t < 4; ++t)
#pragma unroll
      for (int r = 0; r < 8; ++r) o[t][r] *= arw[r];

    // ---- P @ V : A-frag from LDS, B-frag strided from vh ----
    v16h pa;
    {
      const _Float16* pp = &Pt[l16][0];
      half8 lo = *(const half8*)(pp + hf * 16);
      half8 hi = *(const half8*)(pp + hf * 16 + 8);
#pragma unroll
      for (int i = 0; i < 8; ++i) { pa[i] = lo[i]; pa[i + 8] = hi[i]; }
    }
#pragma unroll
    for (int t = 0; t < 4; ++t) {
      v16h vb;
      const int d = t * 16 + l16;
#pragma unroll
      for (int j = 0; j < 16; ++j) {
        int kk = hf * 16 + j;
        vb[j] = vh[(rowbase + kbase + kk) * KVD + kcoff + d];
      }
      o[t] = __builtin_amdgcn_wmma_f32_16x16x32_f16(
          false, pa, false, vb, (short)0, o[t], false, false);
    }
    __syncthreads();   // Pt reused next iteration
  }

  // ---- normalize and store (f16 activations for the wo GEMM) ----
  float linv = 1.0f / l_run;               // column layout (q = l16)
  float inv[8];
#pragma unroll
  for (int r = 0; r < 8; ++r) inv[r] = __shfl(linv, hf * 8 + r, 32);
#pragma unroll
  for (int t = 0; t < 4; ++t)
#pragma unroll
    for (int r = 0; r < 8; ++r) {
      int qrow = qtile * 16 + r + hf * 8;
      oh[(rowbase + qrow) * DIMV + qcoff + t * 16 + l16] =
          (_Float16)(o[t][r] * inv[r]);
    }
}

// ---------------------------------------------------------------------------
// Host-side orchestration
// ---------------------------------------------------------------------------
extern "C" void kernel_launch(void* const* d_in, const int* in_sizes, int n_in,
                              void* d_out, int out_size, void* d_ws,
                              size_t ws_size, hipStream_t stream) {
  const float* x  = (const float*)d_in[0];
  const float* wq = (const float*)d_in[1];
  const float* wk = (const float*)d_in[2];
  const float* wv = (const float*)d_in[3];
  const float* wo = (const float*)d_in[4];

  const int nX  = in_sizes[0];   // 8388608
  const int nWq = in_sizes[1];   // 4194304
  const int nWk = in_sizes[2];   // 1048576
  const int nWv = in_sizes[3];   // 1048576
  const int nWo = in_sizes[4];   // 4194304

  // ---- workspace carve-out ----
  char* ws = (char*)d_ws;
  size_t cur = 0;
  auto take = [&](size_t bytes) {
    void* p = ws + cur;
    cur = (cur + bytes + 255) & ~(size_t)255;
    return p;
  };
  float*    partial = (float*)take(4 * 256 * sizeof(float));
  float*    scales  = (float*)take(4 * sizeof(float));
  _Float16* xh  = (_Float16*)take((size_t)nX  * 2);
  _Float16* wqh = (_Float16*)take((size_t)nWq * 2);
  _Float16* wkh = (_Float16*)take((size_t)nWk * 2);
  _Float16* wvh = (_Float16*)take((size_t)nWv * 2);
  _Float16* woh = (_Float16*)take((size_t)nWo * 2);
  _Float16* qh  = (_Float16*)take((size_t)MROWS * DIMV * 2);
  _Float16* kh  = (_Float16*)take((size_t)MROWS * KVD  * 2);
  _Float16* vh  = (_Float16*)take((size_t)MROWS * KVD  * 2);
  _Float16* ah  = (_Float16*)take((size_t)MROWS * DIMV * 2);

  // ---- 1) scales = max(mean|w|, 1e-6), deterministic tree reductions ----
  k_zero_f32<<<5, 256, 0, stream>>>(partial, 4 * 256 + 4);
  const float* wptr[4] = {wq, wk, wv, wo};
  const int    wn[4]   = {nWq, nWk, nWv, nWo};
  for (int i = 0; i < 4; ++i)
    k_abs_sum_partial<<<256, 256, 0, stream>>>(wptr[i], wn[i],
                                               partial + i * 256);
  for (int i = 0; i < 4; ++i)
    k_finalize_scale<<<1, 256, 0, stream>>>(partial + i * 256, 256,
                                            1.0f / (float)wn[i], scales + i);

  // ---- 2) ternarize weights + convert activations to f16 ----
  _Float16* wdst[4] = {wqh, wkh, wvh, woh};
  for (int i = 0; i < 4; ++i)
    k_ternarize_f16<<<2048, 256, 0, stream>>>(wptr[i], wdst[i], wn[i],
                                              scales + i);
  k_cvt_f16<<<4096, 256, 0, stream>>>(x, xh, nX);

  // ---- 3) projections: q = x@Twq^T, k = x@Twk^T, v = x@Twv^T ----
  k_gemm_wmma<false><<<dim3(DIMV / 64, MROWS / 64), 128, 0, stream>>>(
      xh, wqh, qh, MROWS, DIMV, DIMV);
  k_gemm_wmma<false><<<dim3(KVD / 64, MROWS / 64), 128, 0, stream>>>(
      xh, wkh, kh, MROWS, KVD, DIMV);
  k_gemm_wmma<false><<<dim3(KVD / 64, MROWS / 64), 128, 0, stream>>>(
      xh, wvh, vh, MROWS, KVD, DIMV);

  // ---- 4) causal GQA flash attention (one wave per 16-query tile) ----
  k_attn_wmma<<<dim3(SEQ_ / 16, BSZ_ * NHEADS), 32, 0, stream>>>(qh, kh, vh,
                                                                 ah);

  // ---- 5) output projection straight to fp32 d_out ----
  k_gemm_wmma<true><<<dim3(DIMV / 64, MROWS / 64), 128, 0, stream>>>(
      ah, woh, d_out, MROWS, DIMV, DIMV);
}
